// MultiHeadCrossAttention_20761871909530
// MI455X (gfx1250) — compile-verified, matrix operand order checked by reading, not run on hardware
//
#include <hip/hip_runtime.h>
#include <math.h>

#define D_MODEL 512
#define NUM_HEADS 8
#define D_K 64
#define NP_TOK 4096
#define NL_TOK 2048
#define LN_EPS 1e-5f

typedef _Float16 v16h __attribute__((ext_vector_type(16)));
typedef _Float16 v8h  __attribute__((ext_vector_type(8)));
typedef _Float16 v4h  __attribute__((ext_vector_type(4)));
typedef float    v8f  __attribute__((ext_vector_type(8)));

union HF16 { v16h v; v8h h[2]; };

// ---------------------------------------------------------------------------
// Fragment loader per CDNA5 ISA 7.12.2 (16-bit A 16x32 / B 32x16):
//   lanes 0-15 : elems 0..7 = K[kbase+0..7],  elems 8..15 = K[kbase+16..23]
//   lanes 16-31: elems 0..7 = K[kbase+8..15], elems 8..15 = K[kbase+24..31]
// `base` points at this lane's row (A) / column (B), contiguous over K.
// ---------------------------------------------------------------------------
__device__ __forceinline__ v16h load_frag_h16(const _Float16* __restrict__ base,
                                              int kbase, int lane) {
  const int koff = (lane & 16) ? 8 : 0;
  const _Float16* p = base + kbase + koff;
  HF16 u;
  u.h[0] = *(const v8h*)(p);
  u.h[1] = *(const v8h*)(p + 16);
  return u.v;
}

__device__ __forceinline__ v8f wmma16(v16h a, v16h b, v8f c) {
  return __builtin_amdgcn_wmma_f32_16x16x32_f16(false, a, false, b,
                                                (short)0, c, false, false);
}

// ---------------------------------------------------------------------------
// 0) f32 -> f16 conversion, 4 elems/thread
// ---------------------------------------------------------------------------
__global__ void cvt_f32_to_f16_x4(const float4* __restrict__ in,
                                  v4h* __restrict__ out, int n4) {
  int i = blockIdx.x * blockDim.x + threadIdx.x;
  if (i < n4) {
    float4 f = in[i];
    v4h o;
    o[0] = (_Float16)f.x; o[1] = (_Float16)f.y;
    o[2] = (_Float16)f.z; o[3] = (_Float16)f.w;
    out[i] = o;
  }
}

// ---------------------------------------------------------------------------
// 1) Projection GEMM: C[M x 512] = A[M x 512] @ W^T + bias, f16 out.
//    One wave computes a 16x64 strip. All 5 fragments of a k-step are loaded
//    into independent registers BEFORE the 4 WMMAs so the loads form one
//    clause and the WMMA burst overlaps the next iteration's loads.
//    transposed=0 : out[m*512 + n]   |   transposed=1 : out[n*ldo + m]
// ---------------------------------------------------------------------------
__global__ void __launch_bounds__(128)
proj_gemm_f16(const _Float16* __restrict__ A, const _Float16* __restrict__ W,
              const float* __restrict__ bias, _Float16* __restrict__ out,
              int M, float scale, int transposed, int ldo) {
  const int lane = threadIdx.x & 31;
  const int wave = threadIdx.x >> 5;
  const int tile = blockIdx.x * 4 + wave;
  const int total = (M >> 4) * (D_MODEL >> 6);
  if (tile >= total) return;                 // wave-uniform: EXEC stays all-1s
  const int tm = tile >> 3;
  const int tg = tile & 7;                   // 64-col group
  const int rowA = lane & 15;

  const _Float16* arow = A + (size_t)(tm * 16 + rowA) * D_MODEL;
  const _Float16* bcol = W + (size_t)(tg * 64 + rowA) * D_MODEL;  // +j*16*512

  v8f c[4] = {};
#pragma unroll 2
  for (int k = 0; k < D_MODEL; k += 32) {
    __builtin_prefetch(arow + k + 64, 0, 1);
    v16h a  = load_frag_h16(arow, k, lane);
    v16h b0 = load_frag_h16(bcol,                  k, lane);
    v16h b1 = load_frag_h16(bcol + 16 * D_MODEL,   k, lane);
    v16h b2 = load_frag_h16(bcol + 32 * D_MODEL,   k, lane);
    v16h b3 = load_frag_h16(bcol + 48 * D_MODEL,   k, lane);
    c[0] = wmma16(a, b0, c[0]);
    c[1] = wmma16(a, b1, c[1]);
    c[2] = wmma16(a, b2, c[2]);
    c[3] = wmma16(a, b3, c[3]);
  }

  const int mbase = (lane >> 4) << 3;
#pragma unroll
  for (int j = 0; j < 4; ++j) {
    const int n = tg * 64 + j * 16 + (lane & 15);
    const float bv = bias[n];
#pragma unroll
    for (int r = 0; r < 8; ++r) {
      const float v = (c[j][r] + bv) * scale;
      const int m = tm * 16 + mbase + r;
      if (!transposed) out[(size_t)m * D_MODEL + n] = (_Float16)v;
      else             out[(size_t)n * ldo + m]     = (_Float16)v;
    }
  }
}

// ---------------------------------------------------------------------------
// 2) Fused attention: scores -> mask -> softmax -> attn write -> context.
//    One block (4 waves) owns a 16-query strip of one head; the full
//    16x2048 score row lives in LDS (64 KB, enabled by CDNA5's 320KB WGP LDS).
//    attn slab of d_out is written exactly once; context reads LDS only.
// ---------------------------------------------------------------------------
__global__ void __launch_bounds__(128)
attn_fused_kernel(const _Float16* __restrict__ Q, const _Float16* __restrict__ Km,
                  const _Float16* __restrict__ VT,
                  const long long* __restrict__ bp, const long long* __restrict__ bl,
                  float* __restrict__ attn, _Float16* __restrict__ ctx) {
  __shared__ __align__(16) _Float16 sc[16][NL_TOK];   // 64 KB
  __shared__ float redbuf[16][8];
  __shared__ float rowmax[16];
  __shared__ float rowinv[16];

  const int t = threadIdx.x;
  const int lane = t & 31;
  const int wave = t >> 5;
  const int tm = blockIdx.x & 255;           // Np/16 = 256 strips
  const int h  = blockIdx.x >> 8;
  const int rowA  = lane & 15;
  const int mbase = (lane >> 4) << 3;

  // ---- phase 1: masked scores -> LDS (f16, 1/sqrt(dk) folded into Q) ----
  const _Float16* qrow = Q + (size_t)(tm * 16 + rowA) * D_MODEL + h * D_K;
  long long bprow[8];
#pragma unroll
  for (int r = 0; r < 8; ++r) bprow[r] = bp[tm * 16 + mbase + r];

  for (int g = 0; g < 8; ++g) {              // wave owns cols [wave*512, +512)
    const int l0 = wave * 512 + g * 64;
    const _Float16* kcol = Km + (size_t)(l0 + rowA) * D_MODEL + h * D_K;
    v8f c[4] = {};
#pragma unroll
    for (int k = 0; k < D_K; k += 32) {
      v16h a  = load_frag_h16(qrow, k, lane);
      v16h b0 = load_frag_h16(kcol,                k, lane);
      v16h b1 = load_frag_h16(kcol + 16 * D_MODEL, k, lane);
      v16h b2 = load_frag_h16(kcol + 32 * D_MODEL, k, lane);
      v16h b3 = load_frag_h16(kcol + 48 * D_MODEL, k, lane);
      c[0] = wmma16(a, b0, c[0]);
      c[1] = wmma16(a, b1, c[1]);
      c[2] = wmma16(a, b2, c[2]);
      c[3] = wmma16(a, b3, c[3]);
    }
#pragma unroll
    for (int j = 0; j < 4; ++j) {
      const int l = l0 + j * 16 + (lane & 15);
      const long long blv = bl[l];
#pragma unroll
      for (int r = 0; r < 8; ++r)
        sc[mbase + r][l] = (_Float16)((bprow[r] == blv) ? c[j][r] : -30000.0f);
    }
  }
  __syncthreads();

  // ---- phase 2: row max (8 threads per row, v8h chunks) ----
  const int r8  = t >> 3;
  const int seg = t & 7;
  {
    float mx = -3.0e38f;
    const v8h* p = (const v8h*)&sc[r8][seg * 256];
#pragma unroll 4
    for (int i = 0; i < 32; ++i) {
      v8h x = p[i];
#pragma unroll
      for (int e = 0; e < 8; ++e) mx = fmaxf(mx, (float)x[e]);
    }
    redbuf[r8][seg] = mx;
  }
  __syncthreads();
  if (t < 16) {
    float m = redbuf[t][0];
#pragma unroll
    for (int i = 1; i < 8; ++i) m = fmaxf(m, redbuf[t][i]);
    rowmax[t] = m;
  }
  __syncthreads();

  // ---- phase 3: e = exp(s - mx) kept unnormalized in LDS; row sums ----
  {
    const float mx = rowmax[r8];
    float sum = 0.f;
    v8h* p = (v8h*)&sc[r8][seg * 256];
#pragma unroll 4
    for (int i = 0; i < 32; ++i) {
      v8h x = p[i];
#pragma unroll
      for (int e = 0; e < 8; ++e) {
        const float ev = __expf((float)x[e] - mx);
        sum += ev;
        x[e] = (_Float16)ev;
      }
      p[i] = x;
    }
    redbuf[r8][seg] = sum;
  }
  __syncthreads();
  if (t < 16) {
    float s = redbuf[t][0];
#pragma unroll
    for (int i = 1; i < 8; ++i) s += redbuf[t][i];
    rowinv[t] = 1.f / s;
  }
  __syncthreads();

  // ---- phase 4: normalized attn -> global (coalesced float4, single write) --
  float* arow0 = attn + ((size_t)h * NP_TOK + tm * 16) * NL_TOK;
  for (int r = 0; r < 16; ++r) {
    const float inv = rowinv[r];
    float* orow = arow0 + (size_t)r * NL_TOK;
#pragma unroll
    for (int cc = 0; cc < 4; ++cc) {
      const int col = (cc * 128 + t) * 4;
      v4h x = *(const v4h*)&sc[r][col];
      float4 o;
      o.x = (float)x[0] * inv; o.y = (float)x[1] * inv;
      o.z = (float)x[2] * inv; o.w = (float)x[3] * inv;
      *(float4*)&orow[col] = o;
    }
  }

  // ---- phase 5: context = (e from LDS) @ V^T-cols; scale by rowinv ----
  const int d = h * D_K + wave * 16 + (lane & 15);   // 4 waves cover dk=64
  const _Float16* bcol = VT + (size_t)d * NL_TOK;
  v8f cacc = {};
  const int koff = (lane & 16) ? 8 : 0;
#pragma unroll 4
  for (int k = 0; k < NL_TOK; k += 32) {
    __builtin_prefetch(bcol + k + 64, 0, 1);
    v16h b = load_frag_h16(bcol, k, lane);   // global load first
    HF16 ua;                                  // then LDS reads
    ua.h[0] = *(const v8h*)&sc[rowA][k + koff];
    ua.h[1] = *(const v8h*)&sc[rowA][k + koff + 16];
    cacc = wmma16(ua.v, b, cacc);
  }
#pragma unroll
  for (int r = 0; r < 8; ++r) {
    const float v = cacc[r] * rowinv[mbase + r];
    ctx[(size_t)(tm * 16 + mbase + r) * D_MODEL + d] = (_Float16)v;
  }
}

// ---------------------------------------------------------------------------
// 3) Output projection + residual: y = protein + ctx @ Wo^T + bo (f32 out)
//    16x64 per wave, load-then-compute per k-step.
// ---------------------------------------------------------------------------
__global__ void __launch_bounds__(128)
outproj_kernel(const _Float16* __restrict__ ctx, const _Float16* __restrict__ Wo,
               const float* __restrict__ bo, const float* __restrict__ resid,
               float* __restrict__ y) {
  const int lane = threadIdx.x & 31;
  const int wave = threadIdx.x >> 5;
  const int tile = blockIdx.x * 4 + wave;    // 256 * 8 tiles
  const int tm = tile >> 3;
  const int tg = tile & 7;
  const int rowA = lane & 15;

  const _Float16* arow = ctx + (size_t)(tm * 16 + rowA) * D_MODEL;
  const _Float16* bcol = Wo  + (size_t)(tg * 64 + rowA) * D_MODEL;

  v8f c[4] = {};
#pragma unroll 2
  for (int k = 0; k < D_MODEL; k += 32) {
    __builtin_prefetch(arow + k + 64, 0, 1);
    v16h a  = load_frag_h16(arow, k, lane);
    v16h b0 = load_frag_h16(bcol,                k, lane);
    v16h b1 = load_frag_h16(bcol + 16 * D_MODEL, k, lane);
    v16h b2 = load_frag_h16(bcol + 32 * D_MODEL, k, lane);
    v16h b3 = load_frag_h16(bcol + 48 * D_MODEL, k, lane);
    c[0] = wmma16(a, b0, c[0]);
    c[1] = wmma16(a, b1, c[1]);
    c[2] = wmma16(a, b2, c[2]);
    c[3] = wmma16(a, b3, c[3]);
  }

  const int mbase = (lane >> 4) << 3;
#pragma unroll
  for (int j = 0; j < 4; ++j) {
    const int n = tg * 64 + j * 16 + (lane & 15);
    const float bv = bo[n];
#pragma unroll
    for (int r = 0; r < 8; ++r) {
      const size_t idx = (size_t)(tm * 16 + mbase + r) * D_MODEL + n;
      y[idx] = c[j][r] + bv + resid[idx];
    }
  }
}

// ---------------------------------------------------------------------------
// 4) Row LayerNorm over 512 (one block per protein token)
// ---------------------------------------------------------------------------
__global__ void __launch_bounds__(256)
layernorm_kernel(const float* __restrict__ y, const float* __restrict__ gamma,
                 const float* __restrict__ beta, float* __restrict__ out) {
  __shared__ float red[256];
  const int t = threadIdx.x;
  const float* __restrict__ x = y + (size_t)blockIdx.x * D_MODEL;
  const float a = x[t], b = x[t + 256];

  red[t] = a + b; __syncthreads();
  for (int s = 128; s > 0; s >>= 1) {
    if (t < s) red[t] += red[t + s];
    __syncthreads();
  }
  const float mu = red[0] * (1.f / D_MODEL); __syncthreads();

  const float da = a - mu, db = b - mu;
  red[t] = da * da + db * db; __syncthreads();
  for (int s = 128; s > 0; s >>= 1) {
    if (t < s) red[t] += red[t + s];
    __syncthreads();
  }
  const float inv = rsqrtf(red[0] * (1.f / D_MODEL) + LN_EPS);

  float* __restrict__ o = out + (size_t)blockIdx.x * D_MODEL;
  o[t]       = da * inv * gamma[t]       + beta[t];
  o[t + 256] = db * inv * gamma[t + 256] + beta[t + 256];
}

// ---------------------------------------------------------------------------
extern "C" void kernel_launch(void* const* d_in, const int* in_sizes, int n_in,
                              void* d_out, int out_size, void* d_ws, size_t ws_size,
                              hipStream_t stream) {
  (void)in_sizes; (void)n_in; (void)out_size; (void)ws_size;

  const float*     protein = (const float*)d_in[0];
  const float*     ligand  = (const float*)d_in[1];
  const long long* batch_p = (const long long*)d_in[2];
  const long long* batch_l = (const long long*)d_in[3];
  const float* Wq = (const float*)d_in[4];
  const float* bq = (const float*)d_in[5];
  const float* Wk = (const float*)d_in[6];
  const float* bk = (const float*)d_in[7];
  const float* Wv = (const float*)d_in[8];
  const float* bv = (const float*)d_in[9];
  const float* Wo = (const float*)d_in[10];
  const float* bo = (const float*)d_in[11];
  const float* ln_gamma = (const float*)d_in[12];
  const float* ln_beta  = (const float*)d_in[13];

  float* out  = (float*)d_out;                       // [4096, 512]
  float* attn = out + (size_t)NP_TOK * D_MODEL;      // [8, 4096, 2048]

  // workspace carve-out (256B aligned slices)
  char* w = (char*)d_ws;
  auto take = [&](size_t bytes) -> void* {
    void* p = (void*)w;
    w += (bytes + 255) & ~(size_t)255;
    return p;
  };
  _Float16* prot16 = (_Float16*)take((size_t)NP_TOK * D_MODEL * 2);
  _Float16* lig16  = (_Float16*)take((size_t)NL_TOK * D_MODEL * 2);
  _Float16* wq16   = (_Float16*)take((size_t)D_MODEL * D_MODEL * 2);
  _Float16* wk16   = (_Float16*)take((size_t)D_MODEL * D_MODEL * 2);
  _Float16* wv16   = (_Float16*)take((size_t)D_MODEL * D_MODEL * 2);
  _Float16* wo16   = (_Float16*)take((size_t)D_MODEL * D_MODEL * 2);
  _Float16* q16    = (_Float16*)take((size_t)NP_TOK * D_MODEL * 2);  // * 1/8
  _Float16* k16    = (_Float16*)take((size_t)NL_TOK * D_MODEL * 2);
  _Float16* vt16   = (_Float16*)take((size_t)D_MODEL * NL_TOK * 2);  // [512][2048]
  _Float16* ctx16  = (_Float16*)take((size_t)NP_TOK * D_MODEL * 2);
  float*    y32    = (float*)   take((size_t)NP_TOK * D_MODEL * 4);

  // 0) conversions (float4 -> 4x f16)
  auto cvt = [&](const float* src, _Float16* dst, int n) {
    const int n4 = n / 4;
    cvt_f32_to_f16_x4<<<(n4 + 255) / 256, 256, 0, stream>>>(
        (const float4*)src, (v4h*)dst, n4);
  };
  cvt(protein, prot16, NP_TOK * D_MODEL);
  cvt(ligand,  lig16,  NL_TOK * D_MODEL);
  cvt(Wq, wq16, D_MODEL * D_MODEL);
  cvt(Wk, wk16, D_MODEL * D_MODEL);
  cvt(Wv, wv16, D_MODEL * D_MODEL);
  cvt(Wo, wo16, D_MODEL * D_MODEL);

  // 1) projections: Q (1/sqrt(64) folded), K, V (stored transposed)
  proj_gemm_f16<<<(NP_TOK / 16) * (D_MODEL / 64) / 4, 128, 0, stream>>>(
      prot16, wq16, bq, q16, NP_TOK, 0.125f, 0, 0);
  proj_gemm_f16<<<(NL_TOK / 16) * (D_MODEL / 64) / 4, 128, 0, stream>>>(
      lig16, wk16, bk, k16, NL_TOK, 1.0f, 0, 0);
  proj_gemm_f16<<<(NL_TOK / 16) * (D_MODEL / 64) / 4, 128, 0, stream>>>(
      lig16, wv16, bv, vt16, NL_TOK, 1.0f, 1, NL_TOK);

  // 2) fused scores+softmax+attn-write+context (attn HBM traffic: 1 write)
  attn_fused_kernel<<<NUM_HEADS * (NP_TOK / 16), 128, 0, stream>>>(
      q16, k16, vt16, batch_p, batch_l, attn, ctx16);

  // 3) output projection + residual
  outproj_kernel<<<(NP_TOK / 16) * (D_MODEL / 64) / 4, 128, 0, stream>>>(
      ctx16, wo16, bo, protein, y32);

  // 4) layernorm -> d_out
  layernorm_kernel<<<NP_TOK, 256, 0, stream>>>(y32, ln_gamma, ln_beta, out);
}
